// GNN_LinearAttn_90761248899397
// MI455X (gfx1250) — compile-verified, matrix operand order checked by reading, not run on hardware
//
#include <hip/hip_runtime.h>
#include <hip/hip_bf16.h>
#include <math.h>

#define BATCH 8
#define NN    2048
#define DD    128

typedef __attribute__((ext_vector_type(16))) _Float16 v16h;
typedef __attribute__((ext_vector_type(8)))  _Float16 v8h;
typedef __attribute__((ext_vector_type(8)))  float    v8f;

// ---------------------------------------------------------------------------
// Load a 16x32 f16 WMMA operand tile from a row-major array with leading dim
// `ld` (in elements). This produces the A-operand layout (lane = row, K split
// into two contiguous 8-element chunks per lane), which is ALSO the correct
// B-operand gather when the logical B matrix is the transpose of the
// row-major source (true for every GEMM in this problem: X*W^T, qk*qk^T,
// attn*x via the pre-transposed xT).
//   lane l: row r = l&15, group g = l>>4
//   elements 0..7  -> K = g*8 + 0..7
//   elements 8..15 -> K = 16 + g*8 + 0..7
// ---------------------------------------------------------------------------
__device__ __forceinline__ v16h load_ab_tile(const _Float16* __restrict__ src, int ld) {
  const int lane = threadIdx.x & 31;
  const int r = lane & 15;
  const int g = lane >> 4;
  const _Float16* p = src + (size_t)r * ld + g * 8;
  v8h c0 = *(const v8h*)(p);
  v8h c1 = *(const v8h*)(p + 16);
  v16h out;
#pragma unroll
  for (int e = 0; e < 8; ++e) { out[e] = c0[e]; out[8 + e] = c1[e]; }
  return out;
}

__device__ __forceinline__ v8f wmma_f16(v16h a, v16h b, v8f c) {
  return __builtin_amdgcn_wmma_f32_16x16x32_f16(false, a, false, b, (short)0, c,
                                                false, false);
}

__device__ __forceinline__ float sum16(float v) {
  // reduce across the 16 lanes that share a row group (wave32)
  v += __shfl_xor(v, 1, 32);
  v += __shfl_xor(v, 2, 32);
  v += __shfl_xor(v, 4, 32);
  v += __shfl_xor(v, 8, 32);
  return v;
}

// ---------------------------------------------------------------------------
// K0: convert x (f32) -> xh (f16 row-major) and xt (f16, transposed per batch)
// grid covers exactly B*N*D = 2097152 threads
// ---------------------------------------------------------------------------
__global__ __launch_bounds__(256) void k0_convert_x(const float* __restrict__ x,
                                                    _Float16* __restrict__ xh,
                                                    _Float16* __restrict__ xt) {
  const int idx = blockIdx.x * 256 + threadIdx.x;
  const int d = idx & 127;
  const int n = (idx >> 7) & 2047;
  const int b = idx >> 18;
  const _Float16 h = (_Float16)x[idx];
  xh[idx] = h;
  xt[((size_t)b * DD + d) * NN + n] = h;
}

// K0w: convert the three 128x128 weight matrices to f16 (grid = 3*16384)
__global__ __launch_bounds__(256) void k0_convert_w(const float* __restrict__ Wqk,
                                                    const float* __restrict__ Wl,
                                                    const float* __restrict__ Wr,
                                                    _Float16* __restrict__ wqkh,
                                                    _Float16* __restrict__ wlh,
                                                    _Float16* __restrict__ wrh) {
  const int idx = blockIdx.x * 256 + threadIdx.x;
  if (idx < 16384)          wqkh[idx]         = (_Float16)Wqk[idx];
  else if (idx < 32768)     wlh[idx - 16384]  = (_Float16)Wl[idx - 16384];
  else                      wrh[idx - 32768]  = (_Float16)Wr[idx - 32768];
}

// ---------------------------------------------------------------------------
// K1: qk = sigmoid(x * Wqk^T + bqk), stored as f16 [B*N, 128]
// one wave per 16-row tile; 1024 waves total
// ---------------------------------------------------------------------------
__global__ __launch_bounds__(256) void k1_qk(const _Float16* __restrict__ xh,
                                             const _Float16* __restrict__ wqkh,
                                             const float* __restrict__ bqk,
                                             _Float16* __restrict__ qkh) {
  const int lane = threadIdx.x & 31;
  const int n = lane & 15, g = lane >> 4;
  const int wave = blockIdx.x * 8 + (threadIdx.x >> 5);
  const size_t row0 = (size_t)wave * 16;

  v16h Ax[4];
#pragma unroll
  for (int kc = 0; kc < 4; ++kc)
    Ax[kc] = load_ab_tile(xh + row0 * DD + kc * 32, DD);

#pragma unroll
  for (int dt = 0; dt < 8; ++dt) {
    v8f c;
#pragma unroll
    for (int v = 0; v < 8; ++v) c[v] = 0.f;
#pragma unroll
    for (int kc = 0; kc < 4; ++kc)
      c = wmma_f16(Ax[kc], load_ab_tile(wqkh + (size_t)(dt * 16) * DD + kc * 32, DD), c);
    const float bv = bqk[dt * 16 + n];
#pragma unroll
    for (int v = 0; v < 8; ++v) {
      float t = c[v] + bv;
      t = 1.0f / (1.0f + __expf(-t));
      qkh[(row0 + g * 8 + v) * DD + dt * 16 + n] = (_Float16)t;
    }
  }
}

// ---------------------------------------------------------------------------
// K2: fused masked attention + output projection + L2-normalize + ReLU.
// One wave per 16-query tile (1024 waves). For each 32-key step:
//   S(16x32) = Q * K^T  (8 WMMAs, K=128)  -> scale, mask by adj, accumulate
//   per-row sum; restage S as f16 A-operand via LDS; agg += S * X (8 WMMAs,
//   B-operand from pre-transposed xT with contiguous loads).
// Row division by (rowsum + 1e-6) is deferred (row-scalar commutes with GEMM).
// ---------------------------------------------------------------------------
__global__ __launch_bounds__(256) void k2_attn(const _Float16* __restrict__ qkh,
                                               const _Float16* __restrict__ xh,
                                               const _Float16* __restrict__ xt,
                                               const float* __restrict__ adj,
                                               const _Float16* __restrict__ wlh,
                                               const _Float16* __restrict__ wrh,
                                               const float* __restrict__ bl,
                                               float* __restrict__ out) {
  __shared__ __align__(16) _Float16 smem[8][16 * 128];  // 4KB per wave
  const int widx = threadIdx.x >> 5;
  const int lane = threadIdx.x & 31;
  const int n = lane & 15, g = lane >> 4;
  const int wave = blockIdx.x * 8 + widx;
  const int b = wave >> 7;           // 128 row-tiles per batch
  const int i0 = (wave & 127) << 4;
  const size_t rowbase = (size_t)b * NN + i0;

  _Float16* sm = smem[widx];
  const _Float16* qkb = qkh + (size_t)b * NN * DD;
  const _Float16* xtb = xt + (size_t)b * DD * NN;
  const float* adjrow = adj + rowbase * NN;

  // Q operand tiles (16 rows x 128 K), kept in registers
  v16h Q[4];
#pragma unroll
  for (int kc = 0; kc < 4; ++kc)
    Q[kc] = load_ab_tile(qkh + rowbase * DD + kc * 32, DD);

  v8f acc[8];
#pragma unroll
  for (int dt = 0; dt < 8; ++dt)
#pragma unroll
    for (int v = 0; v < 8; ++v) acc[dt][v] = 0.f;

  float rsum[8];
#pragma unroll
  for (int v = 0; v < 8; ++v) rsum[v] = 0.f;

  const float scale = 0.08838834764831845f;  // 1/sqrt(128)

  for (int j0 = 0; j0 < NN; j0 += 32) {
    v8f s0, s1;
#pragma unroll
    for (int v = 0; v < 8; ++v) { s0[v] = 0.f; s1[v] = 0.f; }
#pragma unroll
    for (int kc = 0; kc < 4; ++kc) {
      v16h bk0 = load_ab_tile(qkb + (size_t)j0 * DD + kc * 32, DD);
      v16h bk1 = load_ab_tile(qkb + (size_t)(j0 + 16) * DD + kc * 32, DD);
      s0 = wmma_f16(Q[kc], bk0, s0);
      s1 = wmma_f16(Q[kc], bk1, s1);
    }
    // scale + adjacency mask + row-sum accumulation; stage S as f16 [16][32]
#pragma unroll
    for (int v = 0; v < 8; ++v) {
      const int m = g * 8 + v;
      const float a0 = adjrow[(size_t)m * NN + j0 + n];
      const float a1 = adjrow[(size_t)m * NN + j0 + 16 + n];
      const float t0 = s0[v] * scale * a0;
      const float t1 = s1[v] * scale * a1;
      rsum[v] += t0 + t1;
      sm[m * 32 + n]      = (_Float16)t0;
      sm[m * 32 + 16 + n] = (_Float16)t1;
    }
    asm volatile("s_wait_dscnt 0" ::: "memory");  // DS RAW fence (in-order per wave)
    v16h As = load_ab_tile(sm, 32);
#pragma unroll
    for (int dt = 0; dt < 8; ++dt) {
      v16h bx = load_ab_tile(xtb + (size_t)(dt * 16) * NN + j0, NN);
      acc[dt] = wmma_f16(As, bx, acc[dt]);
    }
    asm volatile("s_wait_dscnt 0" ::: "memory");  // keep next-iter stores behind reads
  }

  // finalize row sums, apply attn normalization to acc
#pragma unroll
  for (int v = 0; v < 8; ++v) rsum[v] = 1.0f / (sum16(rsum[v]) + 1e-6f);
#pragma unroll
  for (int dt = 0; dt < 8; ++dt)
#pragma unroll
    for (int v = 0; v < 8; ++v)
      sm[(g * 8 + v) * 128 + dt * 16 + n] = (_Float16)(acc[dt][v] * rsum[v]);
  asm volatile("s_wait_dscnt 0" ::: "memory");

  // out = agg*Wl^T + bl + x*Wr^T
  v16h Aagg[4], Axr[4];
#pragma unroll
  for (int kc = 0; kc < 4; ++kc) {
    Aagg[kc] = load_ab_tile(sm + kc * 32, 128);
    Axr[kc]  = load_ab_tile(xh + rowbase * DD + kc * 32, DD);
  }
  v8f ov[8];
#pragma unroll
  for (int dt = 0; dt < 8; ++dt) {
    const float bv = bl[dt * 16 + n];
    v8f c;
#pragma unroll
    for (int v = 0; v < 8; ++v) c[v] = bv;
#pragma unroll
    for (int kc = 0; kc < 4; ++kc) {
      c = wmma_f16(Aagg[kc], load_ab_tile(wlh + (size_t)(dt * 16) * DD + kc * 32, DD), c);
      c = wmma_f16(Axr[kc],  load_ab_tile(wrh + (size_t)(dt * 16) * DD + kc * 32, DD), c);
    }
    ov[dt] = c;
  }

  // row-wise L2 normalize + ReLU + store
  float ss[8];
#pragma unroll
  for (int v = 0; v < 8; ++v) ss[v] = 0.f;
#pragma unroll
  for (int dt = 0; dt < 8; ++dt)
#pragma unroll
    for (int v = 0; v < 8; ++v) ss[v] += ov[dt][v] * ov[dt][v];
#pragma unroll
  for (int v = 0; v < 8; ++v) ss[v] = 1.0f / fmaxf(sqrtf(sum16(ss[v])), 1e-12f);
#pragma unroll
  for (int dt = 0; dt < 8; ++dt)
#pragma unroll
    for (int v = 0; v < 8; ++v) {
      float y = ov[dt][v] * ss[v];
      out[(rowbase + g * 8 + v) * DD + dt * 16 + n] = fmaxf(y, 0.0f);
    }
}

// ---------------------------------------------------------------------------
extern "C" void kernel_launch(void* const* d_in, const int* in_sizes, int n_in,
                              void* d_out, int out_size, void* d_ws, size_t ws_size,
                              hipStream_t stream) {
  (void)in_sizes; (void)n_in; (void)out_size; (void)ws_size;
  const float* x   = (const float*)d_in[0];
  const float* adj = (const float*)d_in[1];
  const float* Wqk = (const float*)d_in[2];
  const float* bqk = (const float*)d_in[3];
  const float* Wl  = (const float*)d_in[4];
  const float* bl  = (const float*)d_in[5];
  const float* Wr  = (const float*)d_in[6];
  float* out = (float*)d_out;

  char* ws = (char*)d_ws;
  _Float16* xh   = (_Float16*)(ws);                              // 4 MiB
  _Float16* xt   = (_Float16*)(ws + (size_t)4  * 1024 * 1024);   // 4 MiB
  _Float16* qkh  = (_Float16*)(ws + (size_t)8  * 1024 * 1024);   // 4 MiB
  _Float16* wqkh = (_Float16*)(ws + (size_t)12 * 1024 * 1024);   // 32 KiB
  _Float16* wlh  = wqkh + 16384;
  _Float16* wrh  = wlh + 16384;

  k0_convert_x<<<8192, 256, 0, stream>>>(x, xh, xt);
  k0_convert_w<<<192, 256, 0, stream>>>(Wqk, Wl, Wr, wqkh, wlh, wrh);
  k1_qk<<<128, 256, 0, stream>>>(xh, wqkh, bqk, qkh);
  k2_attn<<<128, 256, 0, stream>>>(qkh, xh, xt, adj, wlh, wrh, bl, out);
}
